// NodeEdgeBlock_49976239456507
// MI455X (gfx1250) — matrix operand order repaired and checked by reading, not compiled, not run on hardware
//
#include <hip/hip_runtime.h>
#include <hip/hip_bf16.h>
#include <math.h>

// ---------------------------------------------------------------------------
// Problem dims: BS=4, N=256, DX=256, DE=64, DY=64, NH=8, DF=32
// ---------------------------------------------------------------------------
#define INV_SQRT_DF 0.17677669529663687f   // 1/sqrt(32)

typedef __attribute__((ext_vector_type(16))) __bf16 v16bf;
typedef __attribute__((ext_vector_type(8)))  __bf16 v8bf;
typedef __attribute__((ext_vector_type(4)))  __bf16 v4bf;
typedef __attribute__((ext_vector_type(8)))  float  v8f;
typedef __attribute__((ext_vector_type(4)))  unsigned v4u;
typedef __attribute__((ext_vector_type(8)))  int    v8i;
typedef __attribute__((ext_vector_type(4)))  int    v4i;

#if __has_builtin(__builtin_amdgcn_tensor_load_to_lds) && \
    __has_builtin(__builtin_amdgcn_s_wait_tensorcnt)
#define USE_TDM 1
#else
#define USE_TDM 0
#endif

// ---- workspace layout (float elements) ----
static constexpr size_t OFF_Q      = 0;                   // 1024*256
static constexpr size_t OFF_K      = 262144;
static constexpr size_t OFF_V      = 524288;
static constexpr size_t OFF_XPRE   = 786432;              // 1024*256
static constexpr size_t OFF_YE1    = 1048576;             // 4*256
static constexpr size_t OFF_YE2    = OFF_YE1 + 1024;
static constexpr size_t OFF_YX1    = OFF_YE2 + 1024;
static constexpr size_t OFF_YX2    = OFF_YX1 + 1024;
static constexpr size_t OFF_YY     = OFF_YX2 + 1024;      // 4*64
static constexpr size_t OFF_ESUM   = OFF_YY + 256;        // 4*64
static constexpr size_t OFF_ESUMSQ = OFF_ESUM + 256;
static constexpr size_t OFF_EMIN   = OFF_ESUMSQ + 256;    // uint bits
static constexpr size_t OFF_EMAX   = OFF_EMIN + 256;      // uint bits
static constexpr size_t OFF_XPOOL  = OFF_EMAX + 256;      // 4*1024
static constexpr size_t OFF_ZB     = OFF_XPOOL + 4096;    // 256 zeros (bias)

// ---- ordered-int float mapping for atomic min/max ----
__device__ __forceinline__ unsigned f2ord(float f) {
  unsigned u = __float_as_uint(f);
  return (u & 0x80000000u) ? ~u : (u | 0x80000000u);
}
__device__ __forceinline__ float ord2f(unsigned m) {
  unsigned u = (m & 0x80000000u) ? (m ^ 0x80000000u) : ~m;
  return __uint_as_float(u);
}

// ---------------------------------------------------------------------------
// WMMA helpers: D(16x16,f32) = A(16x32,bf16) x B(32x16,bf16) + C
// ---------------------------------------------------------------------------
__device__ __forceinline__ v8f wmma_bf(v16bf a, v16bf b, v8f c) {
  return __builtin_amdgcn_wmma_f32_16x16x32_bf16(
      false, a, false, b, (short)0, c, false, false);
}

// A fragment from row-major LDS tile (16 x K, leading dim = lda elements).
// Per-lane data = two contiguous 8-element (16B) runs -> 2x ds_load_b128.
__device__ __forceinline__ v16bf load_af(const __bf16* A, int lda, int kOff) {
  const int lane = threadIdx.x & 31;
  const int m    = lane & 15;
  const int kb   = (lane >> 4) * 16;
  const __bf16* p = A + m * lda + kOff + kb;
  v8bf lo = *(const v8bf*)p;          // K[kb .. kb+8)
  v8bf hi = *(const v8bf*)(p + 8);    // K[kb+8 .. kb+16)
  return __builtin_shufflevector(lo, hi, 0, 1, 2, 3, 4, 5, 6, 7,
                                 8, 9, 10, 11, 12, 13, 14, 15);
}

// B fragments pre-staged in fragment-major LDS layout:
//   frag block = 512 elems; lane's 16 values at frag*512 + {0,256} + lane*8
// -> 2x ds_load_b128, lane dword-stride 4 => all 64 banks covered.
__device__ __forceinline__ v16bf load_bf(const __bf16* B, int frag) {
  const int lane = threadIdx.x & 31;
  const __bf16* base = B + frag * 512 + lane * 8;
  v8bf lo = *(const v8bf*)base;
  v8bf hi = *(const v8bf*)(base + 256);
  return __builtin_shufflevector(lo, hi, 0, 1, 2, 3, 4, 5, 6, 7,
                                 8, 9, 10, 11, 12, 13, 14, 15);
}

// Stage row-major f32 KxN into fragment-major bf16 LDS. Each thread handles
// 4 consecutive k for one n: one branch-free calc + one 8B store per quad.
__device__ __forceinline__ void stage_b4(__bf16* __restrict__ dst,
                                         const float* __restrict__ src,
                                         int K, int LOGN, int tid) {
  const int N     = 1 << LOGN;
  const int NT    = N >> 4;
  const int items = (K >> 2) << LOGN;     // (K/4)*N quads
  for (int idx = tid; idx < items; idx += 256) {
    const int n    = idx & (N - 1);
    const int k0   = (idx >> LOGN) << 2;  // 4-aligned k
    const int kc   = k0 >> 5, kr = k0 & 31;
    const int lane = ((kr >> 4) << 4) | (n & 15);
    const int i0   = kr & 15;             // i0&7 in {0,4}: 8B aligned
    const int frag = kc * NT + (n >> 4);
    const int pos  = frag * 512 + ((i0 >> 3) << 8) + lane * 8 + (i0 & 7);
    const float* s = src + (size_t)k0 * N + n;
    v4bf o;
    o[0] = (__bf16)s[0];
    o[1] = (__bf16)s[N];
    o[2] = (__bf16)s[2 * N];
    o[3] = (__bf16)s[3 * N];
    *(v4bf*)(dst + pos) = o;
  }
}

#if USE_TDM
// Tensor Data Mover: async DMA of one contiguous 4KB e-tile (1024 f32) into
// LDS. D# per CDNA5 ISA ch.8: group0 {count,lds_addr,global_addr,type=2},
// group1 {data_size=4B, tensor_dim0=1024, tensor_dim1=1, tile_dim0=1024,
// dim0_stride=1024}; groups 2/3 zero (<=2D tensor). Wave-uniform operands;
// caller must restrict issue to ONE wave (TDM ignores EXEC).
__device__ __forceinline__ void tdm_load_tile(const float* gsrc, void* lds_dst) {
  const unsigned lds_addr = (unsigned)(uintptr_t)lds_dst;   // LDS address taken from the low 32 bits
  const unsigned long long ga = (unsigned long long)(uintptr_t)gsrc;
  v4u g0;
  g0[0] = 1u;                                   // count=1, user mode
  g0[1] = lds_addr;                             // lds_addr  [63:32]
  g0[2] = (unsigned)(ga & 0xffffffffu);         // global_addr [95:64]
  g0[3] = (unsigned)((ga >> 32) & 0x01ffffffu)  // global_addr [120:96]
        | (2u << 30);                           // type=2 (image)
  v8i g1;
  g1[0] = 0x00020000;              // wg_mask=0, data_size=2 (4B)
  g1[1] = (int)(1024u << 16);      // tensor_dim0=1024 (bits 79:48, low half)
  g1[2] = (int)(1u << 16);         // tensor_dim0 hi=0; tensor_dim1=1
  g1[3] = (int)(1024u << 16);      // tile_dim0=1024 (bits 127:112)
  g1[4] = 0;                       // tile_dim1=0, tile_dim2=0 (unused)
  g1[5] = 1024;                    // tensor_dim0_stride=1024 (bits 207:160)
  g1[6] = 0;
  g1[7] = 0;
  v4i gz = {0, 0, 0, 0};
#if __clang_major__ >= 23
  v8i gz8 = {0, 0, 0, 0, 0, 0, 0, 0};
  __builtin_amdgcn_tensor_load_to_lds(g0, g1, gz, gz, gz8, 0);
#else
  __builtin_amdgcn_tensor_load_to_lds(g0, g1, gz, gz, 0);
#endif
}
#endif

// ---------------------------------------------------------------------------
// k_prep: y-projections, y@Wyy, init e-pool atomics + zero-bias slab.
// ---------------------------------------------------------------------------
__global__ void k_prep(const float* __restrict__ y,
                       const float* __restrict__ Wye_add, const float* __restrict__ Wye_mul,
                       const float* __restrict__ Wyx_add, const float* __restrict__ Wyx_mul,
                       const float* __restrict__ Wyy, float* __restrict__ ws) {
  const int tid = blockIdx.x * blockDim.x + threadIdx.x;
  if (tid < 4096) {
    const int mat = tid >> 10;
    const int rem = tid & 1023;
    const int b = rem >> 8, o = rem & 255;
    const float* W = (mat == 0) ? Wye_add : (mat == 1) ? Wye_mul
                   : (mat == 2) ? Wyx_add : Wyx_mul;
    float acc = 0.f;
#pragma unroll 4
    for (int k = 0; k < 64; ++k) acc += y[b * 64 + k] * W[k * 256 + o];
    const size_t off = (mat == 0) ? OFF_YE1 : (mat == 1) ? OFF_YE2
                     : (mat == 2) ? OFF_YX1 : OFF_YX2;
    ws[off + rem] = acc;
  } else if (tid < 4352) {
    const int idx = tid - 4096;
    const int b = idx >> 6, o = idx & 63;
    float acc = 0.f;
#pragma unroll 4
    for (int k = 0; k < 64; ++k) acc += y[b * 64 + k] * Wyy[k * 64 + o];
    ws[OFF_YY + idx] = acc;
  } else if (tid < 4608) {
    const int c = tid - 4352;
    ws[OFF_ESUM + c]   = 0.f;
    ws[OFF_ESUMSQ + c] = 0.f;
    ((unsigned*)ws)[OFF_EMIN + c] = 0xFFFFFFFFu;
    ((unsigned*)ws)[OFF_EMAX + c] = 0u;
  } else if (tid < 4864) {
    ws[OFF_ZB + (tid - 4608)] = 0.f;
  }
}

// ---------------------------------------------------------------------------
// k_gemm: C(1024x256) = A(1024x256) @ W(256x256) + bias, * rowmask
// ---------------------------------------------------------------------------
__global__ void __launch_bounds__(256)
k_gemm(const float* __restrict__ A, const float* __restrict__ W,
       const float* __restrict__ bias, const float* __restrict__ rowmask,
       float* __restrict__ C) {
  __shared__ __align__(16) __bf16 As[16 * 264];   // padded stride (264)
  __shared__ __align__(16) __bf16 Ws[16 * 512];   // 16 frags of one k-chunk
  const int tid = threadIdx.x;
  const int m0  = blockIdx.x * 16;

  for (int idx = tid; idx < 16 * 64; idx += 256) {
    const int row = idx >> 6, q = (idx & 63) << 2;
    const float4 f = *(const float4*)&A[(size_t)(m0 + row) * 256 + q];
    v4bf o; o[0] = (__bf16)f.x; o[1] = (__bf16)f.y;
            o[2] = (__bf16)f.z; o[3] = (__bf16)f.w;
    *(v4bf*)&As[row * 264 + q] = o;
  }

  const int w = tid >> 5;
  v8f acc0 = {}; v8f acc1 = {};
  for (int kc = 0; kc < 8; ++kc) {
    __syncthreads();
    stage_b4(Ws, W + (size_t)kc * 32 * 256, 32, 8, tid);
    __syncthreads();
    v16bf a = load_af(As, 264, kc * 32);
    acc0 = wmma_bf(a, load_bf(Ws, 2 * w), acc0);
    acc1 = wmma_bf(a, load_bf(Ws, 2 * w + 1), acc1);
  }
  const int lane = tid & 31;
  const int n = lane & 15, mb = (lane >> 4) * 8;
#pragma unroll
  for (int r = 0; r < 8; ++r) {
    const int row = m0 + mb + r;
    const float rm = rowmask[row];
    const int c0 = w * 32 + n, c1 = c0 + 16;
    C[(size_t)row * 256 + c0] = (acc0[r] + bias[c0]) * rm;
    C[(size_t)row * 256 + c1] = (acc1[r] + bias[c1]) * rm;
  }
}

// ---------------------------------------------------------------------------
// k_edge: fused per-edge pipeline. One workgroup per (b,i), 256 thr = 8 waves.
// e tiles arrive via TDM double-buffering (async DMA overlapped with WMMA).
// ---------------------------------------------------------------------------
__global__ void __launch_bounds__(256)
k_edge(const float* __restrict__ e, const float* __restrict__ node_mask,
       const float* __restrict__ We_mul, const float* __restrict__ We_add,
       const float* __restrict__ We_out, const float* __restrict__ be_out,
       float* __restrict__ ws, float* __restrict__ newE) {
  __shared__ __align__(16) __bf16 Wm_s[32 * 512];   // 32 KB, frag-major (K=64,N=256)
  __shared__ __align__(16) __bf16 Wa_s[32 * 512];   // 32 KB
  __shared__ __align__(16) __bf16 Wo_s[32 * 512];   // 32 KB, frag-major (K=256,N=64)
  __shared__ __align__(16) __bf16 e_s[16 * 72];     // padded row-major 16x64 bf16
  __shared__ __align__(16) __bf16 pre_s[16 * 264];  // padded row-major 16x256
#if USE_TDM
  __shared__ __align__(16) float  e_raw[2][1024];   // TDM double buffer (2x4KB)
#endif
  __shared__ float  mj_s[16];
  __shared__ float  red_s[256];

  const int tid = threadIdx.x;
  const int b = blockIdx.x >> 8;
  const int i = blockIdx.x & 255;
  const float mask_i = node_mask[b * 256 + i];
  const size_t e_base = ((size_t)(b * 256 + i)) * 256 * 64;

#if USE_TDM
  if (tid < 32) tdm_load_tile(e + e_base, &e_raw[0][0]);   // tile 0 in flight
#endif

  stage_b4(Wm_s, We_mul, 64, 8, tid);   // frag = kc*16 + nt, kc<2
  stage_b4(Wa_s, We_add, 64, 8, tid);
  stage_b4(Wo_s, We_out, 256, 6, tid);  // frag = kc*4 + nt, kc<8

  const int w = tid >> 5, lane = tid & 31;
  const int n = lane & 15, mb = (lane >> 4) * 8;

  const float* q_ws = ws + OFF_Q;
  const float* k_ws = ws + OFF_K;
  const float* v_ws = ws + OFF_V;

  float qv[2], ye1v[2], ye2v[2];
  int hd_[2];
#pragma unroll
  for (int t = 0; t < 2; ++t) {
    const int hd = w * 32 + t * 16 + n;
    hd_[t]  = hd;
    qv[t]   = q_ws[((size_t)(b * 256 + i)) * 256 + hd] * INV_SQRT_DF;
    ye1v[t] = ws[OFF_YE1 + b * 256 + hd];
    ye2v[t] = ws[OFF_YE2 + b * 256 + hd];
  }
  float m_run[2] = {-INFINITY, -INFINITY};
  float l_run[2] = {0.f, 0.f};
  float a_run[2] = {0.f, 0.f};

  // e-pool per-thread partials: fixed channel per thread
  const int pc = tid & 63, pjr = tid >> 6;
  float psum = 0.f, psumsq = 0.f, pmin = INFINITY, pmax = -INFINITY;

  for (int jt = 0; jt < 16; ++jt) {
    const int j0 = jt * 16;
#if USE_TDM
    if (tid < 32) __builtin_amdgcn_s_wait_tensorcnt(0);  // tile jt landed
#endif
    __syncthreads();  // prior tile consumed; e_raw[jt&1] visible to all
#if USE_TDM
    if (tid < 32 && jt + 1 < 16)   // kick next DMA; overlaps with compute below
      tdm_load_tile(e + e_base + (size_t)(jt + 1) * 1024, &e_raw[(jt + 1) & 1][0]);
#endif
    if (tid < 16) mj_s[tid] = node_mask[b * 256 + j0 + tid];
#if USE_TDM
    const float* esrc = &e_raw[jt & 1][0];
#pragma unroll
    for (int rr = 0; rr < 4; ++rr) {
      const int row = pjr + rr * 4;
      const float ev = esrc[row * 64 + pc];
      e_s[row * 72 + pc] = (__bf16)ev;
      psum += ev; psumsq += ev * ev;
      pmin = fminf(pmin, ev); pmax = fmaxf(pmax, ev);
    }
#else
#pragma unroll
    for (int rr = 0; rr < 4; ++rr) {
      const int row = pjr + rr * 4;
      const float ev = e[e_base + (size_t)(j0 + row) * 64 + pc];
      e_s[row * 72 + pc] = (__bf16)ev;
      psum += ev; psumsq += ev * ev;
      pmin = fminf(pmin, ev); pmax = fmaxf(pmax, ev);
    }
    if (jt + 1 < 16)
      __builtin_prefetch(&e[e_base + (size_t)(j0 + 16 + pjr * 4) * 64 + pc], 0, 0);
#endif
    __syncthreads();

    // ---- GEMM1 (e1,e2) + elementwise Y/newE_pre + online softmax ----
#pragma unroll
    for (int t = 0; t < 2; ++t) {
      const int nt = 2 * w + t;
      v8f e1 = {}; v8f e2 = {};
#pragma unroll
      for (int kc = 0; kc < 2; ++kc) {
        v16bf a = load_af(e_s, 72, kc * 32);
        e1 = wmma_bf(a, load_bf(Wm_s, kc * 16 + nt), e1);
        e2 = wmma_bf(a, load_bf(Wa_s, kc * 16 + nt), e2);
      }
      const int hd = hd_[t];
      float mx = -INFINITY;
      float Ysm[8];
#pragma unroll
      for (int r = 0; r < 8; ++r) {
        const int j = mb + r;
        const float mj = mj_s[j];
        const float em = mask_i * mj;
        const float kv = k_ws[((size_t)(b * 256 + j0 + j)) * 256 + hd];
        const float Y  = qv[t] * kv * (e1[r] * em + 1.f) + e2[r] * em;
        const float pre = ye1v[t] + (ye2v[t] + 1.f) * Y;
        pre_s[j * 264 + hd] = (__bf16)pre;
        const float ysv = (mj > 0.f) ? Y : -1.0e9f;
        Ysm[r] = ysv;
        mx = fmaxf(mx, ysv);
      }
      mx = fmaxf(mx, __shfl_xor(mx, 16, 32));
      const float mnew  = fmaxf(m_run[t], mx);
      const float alpha = __expf(m_run[t] - mnew);
      float ls = 0.f, avs = 0.f;
#pragma unroll
      for (int r = 0; r < 8; ++r) {
        const float p = __expf(Ysm[r] - mnew);
        ls  += p;
        avs += p * v_ws[((size_t)(b * 256 + j0 + mb + r)) * 256 + hd];
      }
      ls  += __shfl_xor(ls, 16, 32);
      avs += __shfl_xor(avs, 16, 32);
      l_run[t] = l_run[t] * alpha + ls;
      a_run[t] = a_run[t] * alpha + avs;
      m_run[t] = mnew;
    }
    __syncthreads();  // pre_s complete

    // ---- GEMM2: newE tile = pre(16x256) @ We_out(256x64) + b, *em ----
    if (w < 4) {
      v8f d = {};
#pragma unroll
      for (int kc = 0; kc < 8; ++kc) {
        v16bf a = load_af(pre_s, 264, kc * 32);
        d = wmma_bf(a, load_bf(Wo_s, kc * 4 + w), d);
      }
      const int c = w * 16 + n;
      const float bo = be_out[c];
#pragma unroll
      for (int r = 0; r < 8; ++r) {
        const int j = mb + r;
        const float em = mask_i * mj_s[j];
        newE[e_base + (size_t)(j0 + j) * 64 + c] = (d[r] + bo) * em;
      }
    }
  }
  __syncthreads();

  // ---- newX_pre = yx1 + (yx2+1) * (acc/l) ----
  if (lane < 16) {
#pragma unroll
    for (int t = 0; t < 2; ++t) {
      const int hd = hd_[t];
      const float wv = a_run[t] / l_run[t];
      const float x1 = ws[OFF_YX1 + b * 256 + hd];
      const float x2 = ws[OFF_YX2 + b * 256 + hd];
      ws[OFF_XPRE + ((size_t)(b * 256 + i)) * 256 + hd] = x1 + (x2 + 1.f) * wv;
    }
  }

  // ---- e-pool: 256 partials -> 64 channels -> global atomics ----
  red_s[tid] = psum;  __syncthreads();
  if (tid < 64) {
    float s = red_s[tid] + red_s[tid + 64] + red_s[tid + 128] + red_s[tid + 192];
    atomicAdd(&ws[OFF_ESUM + b * 64 + tid], s);
  }
  __syncthreads();
  red_s[tid] = psumsq;  __syncthreads();
  if (tid < 64) {
    float s = red_s[tid] + red_s[tid + 64] + red_s[tid + 128] + red_s[tid + 192];
    atomicAdd(&ws[OFF_ESUMSQ + b * 64 + tid], s);
  }
  __syncthreads();
  red_s[tid] = pmin;  __syncthreads();
  if (tid < 64) {
    float mn = fminf(fminf(red_s[tid], red_s[tid + 64]),
                     fminf(red_s[tid + 128], red_s[tid + 192]));
    atomicMin((unsigned*)ws + OFF_EMIN + b * 64 + tid, f2ord(mn));
  }
  __syncthreads();
  red_s[tid] = pmax;  __syncthreads();
  if (tid < 64) {
    float mx = fmaxf(fmaxf(red_s[tid], red_s[tid + 64]),
                     fmaxf(red_s[tid + 128], red_s[tid + 192]));
    atomicMax((unsigned*)ws + OFF_EMAX + b * 64 + tid, f2ord(mx));
  }
}

// ---------------------------------------------------------------------------
// k_xpool: pool x over axis 1 -> (4,1024) [mean|min|max|std]. 128 blocks.
// ---------------------------------------------------------------------------
__global__ void __launch_bounds__(256)
k_xpool(const float* __restrict__ x, float* __restrict__ ws) {
  const int wg   = blockIdx.x * 8 + (threadIdx.x >> 5);
  const int lane = threadIdx.x & 31;
  const int b = wg >> 8, c = wg & 255;
  float s = 0.f, s2 = 0.f, mn = INFINITY, mx = -INFINITY;
  for (int r = lane; r < 256; r += 32) {
    const float v = x[((size_t)(b * 256 + r)) * 256 + c];
    s += v; s2 += v * v; mn = fminf(mn, v); mx = fmaxf(mx, v);
  }
#pragma unroll
  for (int o = 16; o > 0; o >>= 1) {
    s  += __shfl_xor(s, o, 32);
    s2 += __shfl_xor(s2, o, 32);
    mn  = fminf(mn, __shfl_xor(mn, o, 32));
    mx  = fmaxf(mx, __shfl_xor(mx, o, 32));
  }
  if (lane == 0) {
    float* xp = ws + OFF_XPOOL + (size_t)b * 1024;
    const float var = (s2 - s * s * (1.f / 256.f)) * (1.f / 255.f);
    xp[c]       = s * (1.f / 256.f);
    xp[256 + c] = mn;
    xp[512 + c] = mx;
    xp[768 + c] = sqrtf(fmaxf(var, 0.f));
  }
}

// ---------------------------------------------------------------------------
// k_ynew: newY = (y@Wyy + e_pool@Wey+bey + x_pool@Wxy+bxy) @ Wy_out + by_out
// ---------------------------------------------------------------------------
__global__ void __launch_bounds__(256)
k_ynew(const float* __restrict__ Wxy, const float* __restrict__ bxy,
       const float* __restrict__ Wey, const float* __restrict__ bey,
       const float* __restrict__ Wy_out, const float* __restrict__ by_out,
       const float* __restrict__ ws, float* __restrict__ newY) {
  __shared__ float t_s[256];
  const int tid = threadIdx.x;
  const int b = tid >> 6, o = tid & 63;
  float acc = ws[OFF_YY + b * 64 + o] + bxy[o] + bey[o];
  const float* xp = ws + OFF_XPOOL + (size_t)b * 1024;
  for (int k = 0; k < 1024; ++k) acc += xp[k] * Wxy[k * 64 + o];
  for (int k = 0; k < 256; ++k) {
    const int kind = k >> 6, ch = k & 63;
    const float sum = ws[OFF_ESUM + b * 64 + ch];
    float val;
    if (kind == 0)      val = sum * (1.f / 65536.f);
    else if (kind == 1) val = ord2f(((const unsigned*)ws)[OFF_EMIN + b * 64 + ch]);
    else if (kind == 2) val = ord2f(((const unsigned*)ws)[OFF_EMAX + b * 64 + ch]);
    else {
      const float sq = ws[OFF_ESUMSQ + b * 64 + ch];
      val = sqrtf(fmaxf((sq - sum * sum * (1.f / 65536.f)) * (1.f / 65535.f), 0.f));
    }
    acc += val * Wey[k * 64 + o];
  }
  t_s[tid] = acc;
  __syncthreads();
  float out = by_out[o];
  for (int k = 0; k < 64; ++k) out += t_s[b * 64 + k] * Wy_out[k * 64 + o];
  newY[tid] = out;
}

// ---------------------------------------------------------------------------
extern "C" void kernel_launch(void* const* d_in, const int* in_sizes, int n_in,
                              void* d_out, int out_size, void* d_ws, size_t ws_size,
                              hipStream_t stream) {
  (void)in_sizes; (void)n_in; (void)out_size; (void)ws_size;
  const float* x         = (const float*)d_in[0];
  const float* e         = (const float*)d_in[1];
  const float* y         = (const float*)d_in[2];
  const float* node_mask = (const float*)d_in[3];
  const float* Wq        = (const float*)d_in[4];
  const float* Wk        = (const float*)d_in[5];
  const float* Wv        = (const float*)d_in[6];
  const float* We_mul    = (const float*)d_in[7];
  const float* We_add    = (const float*)d_in[8];
  const float* Wye_add   = (const float*)d_in[9];
  const float* Wye_mul   = (const float*)d_in[10];
  const float* Wyx_add   = (const float*)d_in[11];
  const float* Wyx_mul   = (const float*)d_in[12];
  const float* Wyy       = (const float*)d_in[13];
  const float* Wxy       = (const float*)d_in[14];
  const float* bxy       = (const float*)d_in[15];
  const float* Wey       = (const float*)d_in[16];
  const float* bey       = (const float*)d_in[17];
  const float* We_out    = (const float*)d_in[18];
  const float* be_out    = (const float*)d_in[19];
  const float* Wx_out    = (const float*)d_in[20];
  const float* bx_out    = (const float*)d_in[21];
  const float* Wy_out    = (const float*)d_in[22];
  const float* by_out    = (const float*)d_in[23];

  float* ws   = (float*)d_ws;
  float* outX = (float*)d_out;                 // 4*256*256
  float* outE = outX + 262144;                 // 4*256*256*64
  float* outY = outE + 16777216;               // 4*64

  k_prep<<<19, 256, 0, stream>>>(y, Wye_add, Wye_mul, Wyx_add, Wyx_mul, Wyy, ws);
  k_gemm<<<64, 256, 0, stream>>>(x, Wq, ws + OFF_ZB, node_mask, ws + OFF_Q);
  k_gemm<<<64, 256, 0, stream>>>(x, Wk, ws + OFF_ZB, node_mask, ws + OFF_K);
  k_gemm<<<64, 256, 0, stream>>>(x, Wv, ws + OFF_ZB, node_mask, ws + OFF_V);
  k_edge<<<1024, 256, 0, stream>>>(e, node_mask, We_mul, We_add, We_out, be_out, ws, outE);
  k_gemm<<<64, 256, 0, stream>>>(ws + OFF_XPRE, Wx_out, bx_out, node_mask, outX);
  k_xpool<<<128, 256, 0, stream>>>(x, ws);
  k_ynew<<<1, 256, 0, stream>>>(Wxy, bxy, Wey, bey, Wy_out, by_out, ws, outY);
}
